// YatPerformerAttention_38147899523538
// MI455X (gfx1250) — matrix operand order repaired and checked by reading, not compiled, not run on hardware
//
#include <hip/hip_runtime.h>
#include <hip/hip_bf16.h>

// ---------------- CDNA5 WMMA plumbing ----------------
typedef __attribute__((ext_vector_type(16))) __bf16 v16bf;
typedef __attribute__((ext_vector_type(8)))  __bf16 v8bf;
typedef __attribute__((ext_vector_type(4)))  __bf16 v4bf;
typedef __attribute__((ext_vector_type(8)))  float  v8f;
typedef __attribute__((ext_vector_type(4)))  float  v4f;

#define DEVINL __device__ __forceinline__

// A-fragment (16x32 bf16, MxK) from LDS tile S[m][k] (K innermost, ld mult of 8).
// Lane l<16: row M=l, K {0..7, 16..23}; lanes 16..31: K {8..15, 24..31}.
// Two contiguous 16B reads -> ds_load_b128 x2.
DEVINL v16bf ldA(const __bf16* S, int ld, int m0, int k0, int lane) {
  int m = m0 + (lane & 15);
  int g = (lane >> 4) & 1;
  const __bf16* p = S + m * ld + k0 + 8 * g;
  v8bf lo = *(const v8bf*)p;          // k = k0+8g   .. +7
  v8bf hi = *(const v8bf*)(p + 16);   // k = k0+16+8g.. +7
  return __builtin_shufflevector(lo, hi, 0,1,2,3,4,5,6,7,8,9,10,11,12,13,14,15);
}

// B-fragment (32x16 bf16, KxN) from LDS tile stored TRANSPOSED: S[n][k].
// Lanes 0..15: K {0..7, 16..23} (col N=lane); lanes 16..31: K {8..15, 24..31}.
// Two contiguous 16B reads -> ds_load_b128 x2.
DEVINL v16bf ldBT(const __bf16* S, int ld, int k0, int n0, int lane) {
  int n = n0 + (lane & 15);
  int g = (lane >> 4) & 1;
  const __bf16* p = S + n * ld + k0 + 8 * g;
  v8bf lo = *(const v8bf*)p;
  v8bf hi = *(const v8bf*)(p + 16);
  return __builtin_shufflevector(lo, hi, 0,1,2,3,4,5,6,7,8,9,10,11,12,13,14,15);
}

DEVINL v8f wmma_bf16(v16bf a, v16bf b, v8f c) {
  return __builtin_amdgcn_wmma_f32_16x16x32_bf16(false, a, false, b, (short)0, c,
                                                 false, false);
}

DEVINL v8f v8f_zero() { v8f z = {0.f,0.f,0.f,0.f,0.f,0.f,0.f,0.f}; return z; }

// ---------------- problem constants ----------------
#define BB 2
#define TT 2048
#define EE 512
#define HH 8
#define DD 64        // head dim
#define FF 1024      // POLY_DIM * NUM_SCALES * NUM_PRF
#define NCH 16       // T / CHUNK
#define CH 128       // chunk
#define EPSV 1e-5f

// ---------------- kernels ----------------

__global__ void cvt_f32_bf16(const float* __restrict__ s, __bf16* __restrict__ d, int n4) {
  int i = blockIdx.x * 256 + threadIdx.x;   // one v4 per thread
  if (i < n4) {
    v4f x = *(const v4f*)(s + (size_t)i * 4);
    v4bf y;
#pragma unroll
    for (int j = 0; j < 4; ++j) y[j] = (__bf16)x[j];
    *(v4bf*)(d + (size_t)i * 4) = y;
  }
}

// C[M,N] = A[M,K] @ B[K,N], bf16 in / f32 out. 128x128 block tile, 8 waves (4x2).
__global__ __launch_bounds__(256)
void gemm_bf16(const __bf16* __restrict__ A, const __bf16* __restrict__ B,
               float* __restrict__ C, int M, int N, int K) {
  __shared__ __bf16 As[128][40];    // [m][k], K innermost
  __shared__ __bf16 BsT[128][40];   // [n][k], transposed at staging
  int tid = threadIdx.x, lane = tid & 31, w = tid >> 5;
  int wm = w >> 1, wn = w & 1;
  int bm0 = blockIdx.x * 128, bn0 = blockIdx.y * 128;
  v8f acc[2][4];
#pragma unroll
  for (int mi = 0; mi < 2; ++mi)
#pragma unroll
    for (int ni = 0; ni < 4; ++ni) acc[mi][ni] = v8f_zero();

  for (int kt = 0; kt < K; kt += 32) {
    {   // A tile: 128x32, vector copy (16B loads + 16B LDS stores)
      int r = tid >> 1, half = tid & 1;
      const v8bf* src = (const v8bf*)(A + (size_t)(bm0 + r) * K + kt + half * 16);
      v8bf* dst = (v8bf*)&As[r][half * 16];
      dst[0] = src[0]; dst[1] = src[1];
    }
    {   // B tile: 32x128, vector global read, transposed LDS scatter
      int r = tid >> 3, c0 = (tid & 7) * 16;
      const __bf16* src = B + (size_t)(kt + r) * N + bn0 + c0;
      v8bf s0 = *(const v8bf*)src;
      v8bf s1 = *(const v8bf*)(src + 8);
#pragma unroll
      for (int j = 0; j < 8; ++j) { BsT[c0 + j][r] = s0[j]; BsT[c0 + 8 + j][r] = s1[j]; }
    }
    if (kt + 32 < K)   // global_prefetch_b8 of next A tile
      __builtin_prefetch(A + (size_t)(bm0 + (tid >> 1)) * K + kt + 32, 0, 0);
    __syncthreads();
#pragma unroll
    for (int mi = 0; mi < 2; ++mi) {
      v16bf a = ldA(&As[0][0], 40, wm * 32 + mi * 16, 0, lane);
#pragma unroll
      for (int ni = 0; ni < 4; ++ni) {
        v16bf bb = ldBT(&BsT[0][0], 40, 0, wn * 64 + ni * 16, lane);
        acc[mi][ni] = wmma_bf16(a, bb, acc[mi][ni]);
      }
    }
    __syncthreads();
  }
  int g = lane >> 4, n16 = lane & 15;
#pragma unroll
  for (int mi = 0; mi < 2; ++mi)
#pragma unroll
    for (int ni = 0; ni < 4; ++ni)
#pragma unroll
      for (int r = 0; r < 8; ++r) {
        int m = bm0 + wm * 32 + mi * 16 + r + 8 * g;
        int n = bn0 + wn * 64 + ni * 16 + n16;
        C[(size_t)m * N + n] = acc[mi][ni][r];
      }
}

// Feature map: one wave per token-head. u[64] -> features[1024] bf16.
__global__ __launch_bounds__(256)
void feature_kernel(const float* __restrict__ u_in, const float* __restrict__ proj,
                    const float* __restrict__ scales, const int* __restrict__ skh,
                    const float* __restrict__ sks, __bf16* __restrict__ outF) {
  __shared__ float su[8][64];
  __shared__ float sef[8][64];
  __shared__ float sa[8][16];
  __shared__ float sb[8][16];
  __shared__ float sts[8][16];
  int tid = threadIdx.x, lane = tid & 31, w = tid >> 5;
  int th = blockIdx.x * 8 + w;            // ((b*T + t)*H + h)
  int h = th & 7;
  int bt = th >> 3;
  const float* u = u_in + (size_t)bt * EE + h * DD;
  float u0 = u[lane], u1 = u[lane + 32];
  float ss = u0 * u0 + u1 * u1;
#pragma unroll
  for (int m = 16; m; m >>= 1) ss += __shfl_xor(ss, m, 32);
  float rn = 1.0f / (sqrtf(ss) + 1e-6f);
  u0 *= rn; u1 *= rn;
  su[w][lane] = u0; su[w][lane + 32] = u1;
  if (lane < 16) { sa[w][lane] = 0.f; sb[w][lane] = 0.f; }
  __syncthreads();
#pragma unroll
  for (int half = 0; half < 2; ++half) {
    int sm = lane + 32 * half;
    const float* pr = proj + sm * 64;
    float z = 0.f;
#pragma unroll
    for (int dd = 0; dd < 64; ++dd) z += pr[dd] * su[w][dd];
    int s = sm >> 4;
    float t = scales[2 * s], c = scales[2 * s + 1];
    sef[w][sm] = expf(sqrtf(2.f * t) * z - 2.f * t) * sqrtf(c * 0.0625f);
  }
#pragma unroll
  for (int half = 0; half < 2; ++half) {
    int dd = lane + 32 * half;
    float un = (half == 0) ? u0 : u1;
    atomicAdd(&sa[w][skh[dd]], sks[dd] * un);
    atomicAdd(&sb[w][skh[64 + dd]], sks[64 + dd] * un);
  }
  __syncthreads();
  if (lane < 16) {   // circular convolution == irfft(rfft*rfft)
    float accv = 0.f;
#pragma unroll
    for (int i = 0; i < 16; ++i) accv += sa[w][i] * sb[w][(lane - i) & 15];
    sts[w][lane] = accv;
  }
  __syncthreads();
  __bf16* o = outF + (size_t)th * FF + lane * 32;
#pragma unroll
  for (int jv = 0; jv < 4; ++jv) {
    v8bf pack;
#pragma unroll
    for (int j = 0; j < 8; ++j) {
      int f = lane * 32 + jv * 8 + j;     // f = p*64 + sm
      pack[j] = (__bf16)(sts[w][f >> 6] * sef[w][f & 63]);
    }
    *(v8bf*)(o + jv * 8) = pack;
  }
}

// kv[b,n,h,f,d] = sum_c fk[c,f]*v[c,d]; kz[b,n,h,f] = sum_c fk[c,f]
__global__ __launch_bounds__(256)
void kv_kernel(const __bf16* __restrict__ fkb, const float* __restrict__ vf,
               float* __restrict__ kv, float* __restrict__ kz) {
  __shared__ __bf16 AfkT[256][40];  // [f][c] : K=c innermost
  __shared__ __bf16 BvT[64][40];    // [d][c]
  int tid = threadIdx.x, lane = tid & 31, w = tid >> 5;
  int h = blockIdx.x & 7, nn = (blockIdx.x >> 3) & 15, b = blockIdx.x >> 7;
  int f0 = blockIdx.y * 256;
  v8f acc[2][4];
#pragma unroll
  for (int mi = 0; mi < 2; ++mi)
#pragma unroll
    for (int ni = 0; ni < 4; ++ni) acc[mi][ni] = v8f_zero();
  float kzacc = 0.f;

  for (int ks = 0; ks < 4; ++ks) {        // c in steps of 32
    int c0 = ks * 32;
    {   // fk tile 32x256: vector read, transposed scatter
      int cl = tid >> 3, fb = (tid & 7) * 32;
      const __bf16* src = fkb +
          (((size_t)(b * TT + nn * CH + c0 + cl) * HH) + h) * FF + f0 + fb;
#pragma unroll
      for (int q = 0; q < 4; ++q) {
        v8bf s = *(const v8bf*)(src + q * 8);
#pragma unroll
        for (int j = 0; j < 8; ++j) AfkT[fb + q * 8 + j][cl] = s[j];
      }
    }
    {   // v tile 32x64 f32 -> bf16, transposed scatter
      int cl = tid >> 3, d0 = (tid & 7) * 8;
      const float* src = vf + (size_t)(b * TT + nn * CH + c0 + cl) * EE + h * DD + d0;
      v4f s0 = *(const v4f*)src;
      v4f s1 = *(const v4f*)(src + 4);
#pragma unroll
      for (int j = 0; j < 4; ++j) { BvT[d0 + j][cl] = (__bf16)s0[j];
                                    BvT[d0 + 4 + j][cl] = (__bf16)s1[j]; }
    }
    __syncthreads();
    v16bf bfrag[4];
#pragma unroll
    for (int ni = 0; ni < 4; ++ni) bfrag[ni] = ldBT(&BvT[0][0], 40, 0, ni * 16, lane);
#pragma unroll
    for (int mi = 0; mi < 2; ++mi) {
      v16bf a = ldA(&AfkT[0][0], 40, w * 32 + mi * 16, 0, lane);
#pragma unroll
      for (int ni = 0; ni < 4; ++ni) acc[mi][ni] = wmma_bf16(a, bfrag[ni], acc[mi][ni]);
    }
#pragma unroll
    for (int cl = 0; cl < 32; ++cl) kzacc += (float)AfkT[tid][cl];  // contiguous
    __syncthreads();
  }
  size_t kvbase = (((size_t)(b * NCH + nn) * HH) + h) * FF * DD;
  int g = lane >> 4, n16 = lane & 15;
#pragma unroll
  for (int mi = 0; mi < 2; ++mi)
#pragma unroll
    for (int ni = 0; ni < 4; ++ni)
#pragma unroll
      for (int r = 0; r < 8; ++r) {
        int f = f0 + w * 32 + mi * 16 + r + 8 * g;
        int dd = ni * 16 + n16;
        kv[kvbase + (size_t)f * DD + dd] = acc[mi][ni][r];
      }
  kz[(((size_t)(b * NCH + nn) * HH) + h) * FF + f0 + tid] = kzacc;
}

// exclusive prefix over chunk axis n (in-place)
__global__ void prefix_kv(float* __restrict__ kv) {
  int id = blockIdx.x * 256 + threadIdx.x;
  int dd = id & 63, f = (id >> 6) & 1023, h = (id >> 16) & 7, b = id >> 19;
  size_t base = ((((size_t)b * NCH) * HH + h) * FF + f) * DD + dd;
  const size_t step = (size_t)HH * FF * DD;
  float run = 0.f;
#pragma unroll
  for (int n = 0; n < NCH; ++n) {
    size_t ix = base + (size_t)n * step;
    float v = kv[ix]; kv[ix] = run; run += v;
  }
}

__global__ void prefix_kz(float* __restrict__ kz) {
  int id = blockIdx.x * 256 + threadIdx.x;
  int f = id & 1023, h = (id >> 10) & 7, b = id >> 13;
  size_t base = (((size_t)b * NCH) * HH + h) * FF + f;
  const size_t step = (size_t)HH * FF;
  float run = 0.f;
#pragma unroll
  for (int n = 0; n < NCH; ++n) {
    size_t ix = base + (size_t)n * step;
    float v = kz[ix]; kz[ix] = run; run += v;
  }
}

// Fused chunk attention: one block per (b,n,h).
__global__ __launch_bounds__(256)
void attn_kernel(const __bf16* __restrict__ fqb, const __bf16* __restrict__ fkb,
                 const float* __restrict__ kvp, const float* __restrict__ kzp,
                 const float* __restrict__ vf, __bf16* __restrict__ ao) {
  __shared__ __align__(16) unsigned char smem[53248];
  __shared__ float denp[256];
  __shared__ float denf[128];
  // loop phase
  __bf16* Qs  = (__bf16*)smem;                 // [128][72] : [c][f]
  __bf16* Ks  = (__bf16*)(smem + 18432);       // [128][72] : [c][f]
  __bf16* PsT = (__bf16*)(smem + 36864);       // [64][72]  : [d][f]
  float*  kzc = (float*)(smem + 46080);        // [64]
  // post phase (LDS reuse)
  __bf16* Sc  = (__bf16*)smem;                 // [128][136] : [c][c']
  __bf16* VsT = (__bf16*)(smem + 34816);       // [64][136]  : [d][c']

  int tid = threadIdx.x, lane = tid & 31, w = tid >> 5;
  int h = blockIdx.x & 7, nn = (blockIdx.x >> 3) & 15, b = blockIdx.x >> 7;

  v8f asc[8], anum[4];
#pragma unroll
  for (int i = 0; i < 8; ++i) asc[i] = v8f_zero();
#pragma unroll
  for (int i = 0; i < 4; ++i) anum[i] = v8f_zero();

  float dpart = 0.f;
  int cden = tid & 127, hden = tid >> 7;
  size_t kvbase = (((size_t)(b * NCH + nn) * HH) + h) * FF * DD;
  size_t kzbase = (((size_t)(b * NCH + nn) * HH) + h) * FF;

  for (int fc = 0; fc < 16; ++fc) {           // F in chunks of 64
    int f0 = fc * 64;
    {   // fq/fk tiles 128x64: full 16B vector copies
      int c = tid >> 1, halfq = tid & 1;
      size_t gidx = (((size_t)(b * TT + nn * CH + c) * HH) + h) * FF + f0 + halfq * 32;
      const v8bf* sq = (const v8bf*)(fqb + gidx);
      const v8bf* sk = (const v8bf*)(fkb + gidx);
      v8bf* dq = (v8bf*)&Qs[c * 72 + halfq * 32];
      v8bf* dk = (v8bf*)&Ks[c * 72 + halfq * 32];
#pragma unroll
      for (int q = 0; q < 4; ++q) { dq[q] = sq[q]; dk[q] = sk[q]; }
    }
    {   // kv_prev tile 64x64 f32 -> bf16 transposed to [d][f]
      int fl = tid >> 2, d0 = (tid & 3) * 16;
      const float* src = kvp + kvbase + (size_t)(f0 + fl) * DD + d0;
#pragma unroll
      for (int q = 0; q < 4; ++q) {
        v4f s = *(const v4f*)(src + q * 4);
#pragma unroll
        for (int j = 0; j < 4; ++j) PsT[(d0 + q * 4 + j) * 72 + fl] = (__bf16)s[j];
      }
    }
    if (tid < 64) kzc[tid] = kzp[kzbase + f0 + tid];
    __syncthreads();
#pragma unroll
    for (int kk = 0; kk < 2; ++kk) {
      int k0 = kk * 32;
      v16bf a = ldA(Qs, 72, w * 16, k0, lane);
#pragma unroll
      for (int nt = 0; nt < 8; ++nt) {        // sc += fq @ fk^T
        v16bf bb = ldBT(Ks, 72, k0, nt * 16, lane);
        asc[nt] = wmma_bf16(a, bb, asc[nt]);
      }
#pragma unroll
      for (int nt = 0; nt < 4; ++nt) {        // num += fq @ kv_prev
        v16bf bb = ldBT(PsT, 72, k0, nt * 16, lane);
        anum[nt] = wmma_bf16(a, bb, anum[nt]);
      }
    }
#pragma unroll
    for (int j = 0; j < 32; ++j) {            // den += fq @ kz_prev (VALU)
      int fl = hden * 32 + j;
      dpart += (float)Qs[cden * 72 + fl] * kzc[fl];
    }
    __syncthreads();
  }
  denp[tid] = dpart;
  __syncthreads();

  // masked sc -> LDS (bf16, A-layout: K=c' innermost), stage v transposed
  int g16 = lane >> 4, n16 = lane & 15;
  int crow_base = w * 16;
#pragma unroll
  for (int nt = 0; nt < 8; ++nt)
#pragma unroll
    for (int r = 0; r < 8; ++r) {
      int cr = crow_base + r + 8 * g16;
      int cc = nt * 16 + n16;
      float vsc = (cc <= cr) ? asc[nt][r] : 0.f;   // causal tril mask
      Sc[cr * 136 + cc] = (__bf16)vsc;
    }
  {
    int c = tid >> 1, d0 = (tid & 1) * 32;
    const float* src = vf + (size_t)(b * TT + nn * CH + c) * EE + h * DD + d0;
#pragma unroll
    for (int q = 0; q < 8; ++q) {
      v4f s = *(const v4f*)(src + q * 4);
#pragma unroll
      for (int j = 0; j < 4; ++j) VsT[(d0 + q * 4 + j) * 136 + c] = (__bf16)s[j];
    }
  }
  __syncthreads();

  if (tid < 128) {   // den += intra-chunk row sums of masked sc
    float dsum = denp[tid] + denp[tid + 128];
    for (int k = 0; k < 128; ++k) dsum += (float)Sc[tid * 136 + k];
    denf[tid] = dsum;
  }
  __syncthreads();

  // num += scm @ v
#pragma unroll
  for (int ks = 0; ks < 4; ++ks) {
    v16bf a = ldA(Sc, 136, w * 16, ks * 32, lane);
#pragma unroll
    for (int nt = 0; nt < 4; ++nt) {
      v16bf bb = ldBT(VsT, 136, ks * 32, nt * 16, lane);
      anum[nt] = wmma_bf16(a, bb, anum[nt]);
    }
  }

  // out = num / (den + eps), bf16 for the final Wo GEMM
#pragma unroll
  for (int nt = 0; nt < 4; ++nt)
#pragma unroll
    for (int r = 0; r < 8; ++r) {
      int cr = crow_base + r + 8 * g16;
      int dd = nt * 16 + n16;
      float val = anum[nt][r] / (denf[cr] + EPSV);
      ao[(size_t)(b * TT + nn * CH + cr) * EE + h * DD + dd] = (__bf16)val;
    }
}

// ---------------- host launcher ----------------
extern "C" void kernel_launch(void* const* d_in, const int* in_sizes, int n_in,
                              void* d_out, int out_size, void* d_ws, size_t ws_size,
                              hipStream_t stream) {
  (void)in_sizes; (void)n_in; (void)out_size; (void)ws_size;
  const float* x      = (const float*)d_in[0];
  const float* Wq     = (const float*)d_in[1];
  const float* Wk     = (const float*)d_in[2];
  const float* Wv     = (const float*)d_in[3];
  const float* Wo     = (const float*)d_in[4];
  const float* proj   = (const float*)d_in[5];
  const float* scales = (const float*)d_in[6];
  const int*   sk_h   = (const int*)d_in[7];
  const float* sk_s   = (const float*)d_in[8];

  char* p = (char*)d_ws;
  auto carve = [&](size_t bytes) { void* r = (void*)p; p += (bytes + 255) & ~(size_t)255; return r; };

  const int BT = BB * TT;                       // 4096
  __bf16* xb  = (__bf16*)carve((size_t)BT * EE * 2);
  __bf16* Wqb = (__bf16*)carve((size_t)EE * EE * 2);
  __bf16* Wkb = (__bf16*)carve((size_t)EE * EE * 2);
  __bf16* Wvb = (__bf16*)carve((size_t)EE * EE * 2);
  __bf16* Wob = (__bf16*)carve((size_t)EE * EE * 2);
  float*  qf  = (float*)carve((size_t)BT * EE * 4);
  float*  kf  = (float*)carve((size_t)BT * EE * 4);
  float*  vfb = (float*)carve((size_t)BT * EE * 4);
  __bf16* fqb = (__bf16*)carve((size_t)BT * HH * FF * 2);
  __bf16* fkb = (__bf16*)carve((size_t)BT * HH * FF * 2);
  float*  kvw = (float*)carve((size_t)BB * NCH * HH * FF * DD * 4);
  float*  kzw = (float*)carve((size_t)BB * NCH * HH * FF * 4);
  __bf16* aob = (__bf16*)carve((size_t)BT * EE * 2);

  // 1) f32 -> bf16 (vectorized x4)
  cvt_f32_bf16<<<(BT * EE / 4 + 255) / 256, 256, 0, stream>>>(x, xb, BT * EE / 4);
  cvt_f32_bf16<<<(EE * EE / 4 + 255) / 256, 256, 0, stream>>>(Wq, Wqb, EE * EE / 4);
  cvt_f32_bf16<<<(EE * EE / 4 + 255) / 256, 256, 0, stream>>>(Wk, Wkb, EE * EE / 4);
  cvt_f32_bf16<<<(EE * EE / 4 + 255) / 256, 256, 0, stream>>>(Wv, Wvb, EE * EE / 4);
  cvt_f32_bf16<<<(EE * EE / 4 + 255) / 256, 256, 0, stream>>>(Wo, Wob, EE * EE / 4);

  // 2) q,k,v projections (WMMA)
  dim3 gg(BT / 128, EE / 128);
  gemm_bf16<<<gg, 256, 0, stream>>>(xb, Wqb, qf, BT, EE, EE);
  gemm_bf16<<<gg, 256, 0, stream>>>(xb, Wkb, kf, BT, EE, EE);
  gemm_bf16<<<gg, 256, 0, stream>>>(xb, Wvb, vfb, BT, EE, EE);

  // 3) feature maps (wave per token-head)
  feature_kernel<<<BT * HH / 8, 256, 0, stream>>>(qf, proj, scales, sk_h, sk_s, fqb);
  feature_kernel<<<BT * HH / 8, 256, 0, stream>>>(kf, proj, scales, sk_h, sk_s, fkb);

  // 4) per-chunk kv / kz (WMMA)
  kv_kernel<<<dim3(BB * NCH * HH, FF / 256), 256, 0, stream>>>(fkb, vfb, kvw, kzw);

  // 5) exclusive prefix over chunks
  prefix_kv<<<(BB * HH * FF * DD) / 256, 256, 0, stream>>>(kvw);
  prefix_kz<<<(BB * HH * FF) / 256, 256, 0, stream>>>(kzw);

  // 6) fused chunk attention (WMMA x3 contractions)
  attn_kernel<<<BB * NCH * HH, 256, 0, stream>>>(fqb, fkb, kvw, kzw, vfb, aob);

  // 7) output projection (WMMA) -> f32 d_out
  gemm_bf16<<<gg, 256, 0, stream>>>(aob, Wob, (float*)d_out, BT, EE, EE);
}
